// MultiHeadAttention_11106785427937
// MI455X (gfx1250) — matrix operand order repaired and checked by reading, not compiled, not run on hardware
//
#include <hip/hip_runtime.h>
#include <cstdint>

typedef __attribute__((ext_vector_type(16))) __bf16 bf16x16;
typedef __attribute__((ext_vector_type(8)))  float  f32x8;

static constexpr int B_  = 4;
static constexpr int S_  = 2048;
static constexpr int D_  = 2048;    // model dim
static constexpr int H_  = 2;       // heads
static constexpr int HD_ = 1024;    // head dim
static constexpr int N3_ = 3 * D_;  // 6144
static constexpr int M_  = B_ * S_; // 8192
static constexpr float SCALE_ = 0.03125f; // 1/sqrt(1024)

__device__ __forceinline__ unsigned short f2bf(float f) {
  unsigned u = __float_as_uint(f);
  u += 0x7FFFu + ((u >> 16) & 1u);          // round-to-nearest-even
  return (unsigned short)(u >> 16);
}
__device__ __forceinline__ float bf2f(unsigned short h) {
  return __uint_as_float(((unsigned)h) << 16);
}

union AFrag { bf16x16 v; uint4 q[2]; unsigned short u[16]; };
union Acc8  { f32x8 v; float f[8]; };

__device__ __forceinline__ f32x8 wmma_bf16(bf16x16 a, bf16x16 b, f32x8 c) {
  return __builtin_amdgcn_wmma_f32_16x16x32_bf16(false, a, false, b, (short)0, c,
                                                 false, false);
}

// ---- CDNA5 async global->LDS copy (ASYNCcnt path) --------------------------
__device__ __forceinline__ unsigned lds_off_of(const void* p) {
  // low 32 bits of a generic LDS address are the raw LDS byte offset
  return (unsigned)(unsigned long long)(uintptr_t)p;
}
__device__ __forceinline__ void async_copy_b128(unsigned lds_off, const void* g) {
  unsigned long long ga = (unsigned long long)(uintptr_t)g;
  asm volatile("global_load_async_to_lds_b128 %0, %1, off"
               :: "v"(lds_off), "v"(ga) : "memory");
}
__device__ __forceinline__ void wait_async0() {
  asm volatile("s_wait_asynccnt 0" ::: "memory");
}

// ============================================================================
// One-time f32 -> bf16 conversion (x, wqkv, wout), 8 elements per thread.
// ============================================================================
__global__ __launch_bounds__(256) void f32_to_bf16_kernel(
    const float* __restrict__ src, unsigned short* __restrict__ dst) {
  const int i = blockIdx.x * 256 + threadIdx.x;
  const float4 f0 = reinterpret_cast<const float4*>(src)[(size_t)i * 2];
  const float4 f1 = reinterpret_cast<const float4*>(src)[(size_t)i * 2 + 1];
  union { uint4 q; unsigned short u[8]; } o;
  o.u[0] = f2bf(f0.x); o.u[1] = f2bf(f0.y); o.u[2] = f2bf(f0.z); o.u[3] = f2bf(f0.w);
  o.u[4] = f2bf(f1.x); o.u[5] = f2bf(f1.y); o.u[6] = f2bf(f1.z); o.u[7] = f2bf(f1.w);
  reinterpret_cast<uint4*>(dst)[i] = o.q;
}

// ============================================================================
// Shared GEMM core: C[128x128 tile] += A(bf16,row-major) * B(bf16,row-major),
// KTILE=32, double-buffered LDS.  A staged via async global->LDS b128;
// B staged transposed ([n][k]) with coalesced b128 reads + ds b16 stores.
// Wave layout: 8 waves, wave (wm,wn) owns rows wm*32 (2 m-tiles) x cols wn*64
// (4 n-tiles).  Fragment layouts per CDNA5 ISA.
// ============================================================================
__device__ __forceinline__ void gemm_core_bf16(
    const unsigned short* __restrict__ A, int lda,
    const unsigned short* __restrict__ Bmat, int ldb,
    unsigned short* As, unsigned short* Bs,   // each [2][128*32]
    int m0, int n0, int K, Acc8 acc[2][4]) {
  const int tid  = threadIdx.x;
  const int lane = tid & 31;
  const int wv   = tid >> 5;
  const int hh   = lane >> 4;
  const int lr   = lane & 15;
  const int wm   = wv >> 1;
  const int wn   = wv & 1;
  const int srow = tid >> 1;
  const int skp  = (tid & 1) * 16;
  const int NT   = K / 32;

  auto stage = [&](int kt, int buf) {
    unsigned short* Ab = As + buf * (128 * 32);
    unsigned short* Bb = Bs + buf * (128 * 32);
    const int k0 = kt * 32;
    // A tile: pure byte move -> async engine
    const unsigned short* gA = A + (size_t)(m0 + srow) * lda + k0 + skp;
    __builtin_prefetch(gA + 32, 0, 0);
    async_copy_b128(lds_off_of(Ab + srow * 32 + skp), gA);
    async_copy_b128(lds_off_of(Ab + srow * 32 + skp + 8), gA + 8);
    // B tile: transpose-stage [n][k]; 2 chunks of 8 per thread
#pragma unroll
    for (int r2 = 0; r2 < 2; ++r2) {
      const int c    = tid + r2 * 256;
      const int krow = c >> 4;        // 0..31
      const int nc   = (c & 15) * 8;  // 0..120
      union { uint4 q; unsigned short u[8]; } t;
      t.q = *reinterpret_cast<const uint4*>(Bmat + (size_t)(k0 + krow) * ldb + n0 + nc);
#pragma unroll
      for (int i = 0; i < 8; ++i) Bb[(nc + i) * 32 + krow] = t.u[i];
    }
  };

  stage(0, 0);
  for (int kt = 0; kt < NT; ++kt) {
    const int cur = kt & 1;
    wait_async0();
    __syncthreads();                       // publish buf[cur]
    if (kt + 1 < NT) stage(kt + 1, cur ^ 1);  // overlap next copy with math

    const unsigned short* Ab = As + cur * (128 * 32);
    const unsigned short* Bb = Bs + cur * (128 * 32);
    AFrag a[2], b[4];
#pragma unroll
    for (int mt = 0; mt < 2; ++mt) {
      const unsigned short* p = Ab + (wm * 32 + mt * 16 + lr) * 32 + hh * 8;
      a[mt].q[0] = *reinterpret_cast<const uint4*>(p);
      a[mt].q[1] = *reinterpret_cast<const uint4*>(p + 16);
    }
#pragma unroll
    for (int nt = 0; nt < 4; ++nt) {
      const unsigned short* p = Bb + (wn * 64 + nt * 16 + lr) * 32 + hh * 8;
      b[nt].q[0] = *reinterpret_cast<const uint4*>(p);
      b[nt].q[1] = *reinterpret_cast<const uint4*>(p + 16);
    }
#pragma unroll
    for (int mt = 0; mt < 2; ++mt)
#pragma unroll
      for (int nt = 0; nt < 4; ++nt)
        acc[mt][nt].v = wmma_bf16(a[mt].v, b[nt].v, acc[mt][nt].v);
  }
}

// ============================================================================
// Kernel 1: QKV projection.  qkv[8192x6144] = xbf * wqkvbf (bf16 WMMA, f32
// acc).  Wave-uniform scatter epilogue: q,k -> (b,h,s,d); v -> (b,h,d,s).
// ============================================================================
__global__ __launch_bounds__(256) void qkv_gemm_kernel(
    const unsigned short* __restrict__ X, const unsigned short* __restrict__ W,
    unsigned short* __restrict__ qb, unsigned short* __restrict__ kb,
    unsigned short* __restrict__ vtb) {
  __shared__ unsigned short As[2 * 128 * 32];
  __shared__ unsigned short Bs[2 * 128 * 32];

  const int tid  = threadIdx.x;
  const int lane = tid & 31;
  const int wv   = tid >> 5;
  const int hh   = lane >> 4;
  const int lr   = lane & 15;
  const int wm   = wv >> 1;
  const int wn   = wv & 1;
  const int n0 = blockIdx.x * 128;
  const int m0 = blockIdx.y * 128;

  Acc8 acc[2][4];
#pragma unroll
  for (int mt = 0; mt < 2; ++mt)
#pragma unroll
    for (int nt = 0; nt < 4; ++nt)
#pragma unroll
      for (int j = 0; j < 8; ++j) acc[mt][nt].f[j] = 0.f;

  gemm_core_bf16(X, D_, W, N3_, As, Bs, m0, n0, D_, acc);

  // wave-uniform routing: which/head constant per 16-wide n-tile,
  // batch constant per block (128-row tile never crosses a 2048-row batch)
  const int bb  = m0 >> 11;
  const int sqb = (m0 & (S_ - 1)) + wm * 32;
#pragma unroll
  for (int mt = 0; mt < 2; ++mt) {
    const int sq0r = sqb + mt * 16 + 8 * hh;
#pragma unroll
    for (int nt = 0; nt < 4; ++nt) {
      const int cbase = n0 + wn * 64 + nt * 16 + lr;
      const int which = cbase >> 11;          // 0=q 1=k 2=v (uniform)
      const int head  = (cbase >> 10) & 1;    // uniform
      const int dd    = cbase & (HD_ - 1);
      unsigned short* base;
      size_t off;
      int stride;
      if (which == 2) {
        base = vtb; stride = 1;
        off = (size_t)((bb * H_ + head) * HD_ + dd) * S_ + sq0r;
      } else {
        base = (which == 0) ? qb : kb; stride = HD_;
        off = (size_t)((bb * H_ + head) * S_ + sq0r) * HD_ + dd;
      }
#pragma unroll
      for (int j = 0; j < 8; ++j)
        base[off + (size_t)j * stride] = f2bf(acc[mt][nt].f[j]);
    }
  }
}

// ============================================================================
// Kernel 2: RoPE on q and k, in place.  h=2,d=1024 pairs (head0,dd)<->(head1,dd)
// with angle t * 10000^(-2*(dd%512)/1024).
// ============================================================================
__global__ __launch_bounds__(256) void rope_kernel(
    unsigned short* __restrict__ qb, unsigned short* __restrict__ kb) {
  const int idx = blockIdx.x * 256 + threadIdx.x;  // over b*s*hd
  const int dd = idx & (HD_ - 1);
  const int sq = (idx >> 10) & (S_ - 1);
  const int bb = idx >> 21;
  const int fi = dd & 511;
  const float inv = __expf(-(float)(2 * fi) * (9.210340371976184f / (float)HD_));
  float sn, cs;
  __sincosf((float)sq * inv, &sn, &cs);
  const size_t i0 = (size_t)((bb * H_ + 0) * S_ + sq) * HD_ + dd;
  const size_t i1 = (size_t)((bb * H_ + 1) * S_ + sq) * HD_ + dd;
  {
    float x1 = bf2f(qb[i0]), x2 = bf2f(qb[i1]);
    qb[i0] = f2bf(x1 * cs - x2 * sn);
    qb[i1] = f2bf(x2 * cs + x1 * sn);
  }
  {
    float x1 = bf2f(kb[i0]), x2 = bf2f(kb[i1]);
    kb[i0] = f2bf(x1 * cs - x2 * sn);
    kb[i1] = f2bf(x2 * cs + x1 * sn);
  }
}

// ============================================================================
// Kernel 3: flash attention.  One workgroup per (bh, 16 query rows).
// 8 waves each own a 128-wide d slice; Q fragments register-resident,
// cross-wave S reduction via ds_add_f32, online softmax, WMMA PV with
// pre-transposed V (global, L2-resident: 192MB L2 >> 100MB K/V).
// ============================================================================
__global__ __launch_bounds__(256) void flash_attn_kernel(
    const unsigned short* __restrict__ qb, const unsigned short* __restrict__ kb,
    const unsigned short* __restrict__ vtb, unsigned short* __restrict__ attn) {
  __shared__ float Sbuf[16 * 32];
  __shared__ unsigned short Pb[16 * 32];
  __shared__ float alpha_s[16];
  __shared__ float invl_s[16];

  const int tid  = threadIdx.x;
  const int lane = tid & 31;
  const int wv   = tid >> 5;   // wave owns d-cols [wv*128, wv*128+128)
  const int hh   = lane >> 4;
  const int lr   = lane & 15;
  const int bh   = blockIdx.y;
  const int sq0  = blockIdx.x * 16;

  AFrag qf[4];
#pragma unroll
  for (int c = 0; c < 4; ++c) {
    const size_t base =
        (size_t)(bh * S_ + sq0 + lr) * HD_ + (wv * 4 + c) * 32 + hh * 8;
    qf[c].q[0] = *reinterpret_cast<const uint4*>(qb + base);
    qf[c].q[1] = *reinterpret_cast<const uint4*>(qb + base + 16);
  }

  Acc8 o[8];
#pragma unroll
  for (int nt = 0; nt < 8; ++nt)
#pragma unroll
    for (int j = 0; j < 8; ++j) o[nt].f[j] = 0.f;

  float m_i = -3.4e38f, l_i = 0.f;   // live in threads 0..15

  for (int j0 = 0; j0 < S_; j0 += 32) {
    Sbuf[tid] = 0.f;
    Sbuf[tid + 256] = 0.f;
    __syncthreads();

#pragma unroll
    for (int nt = 0; nt < 2; ++nt) {
      Acc8 s;
#pragma unroll
      for (int j = 0; j < 8; ++j) s.f[j] = 0.f;
#pragma unroll
      for (int c = 0; c < 4; ++c) {
        AFrag bfr;
        const size_t base = (size_t)(bh * S_ + j0 + nt * 16 + lr) * HD_ +
                            (wv * 4 + c) * 32 + hh * 8;
        bfr.q[0] = *reinterpret_cast<const uint4*>(kb + base);
        bfr.q[1] = *reinterpret_cast<const uint4*>(kb + base + 16);
        s.v = wmma_bf16(qf[c].v, bfr.v, s.v);
      }
#pragma unroll
      for (int j = 0; j < 8; ++j)
        atomicAdd(&Sbuf[(j + 8 * hh) * 32 + nt * 16 + lr], s.f[j]);
    }
    __syncthreads();

    if (tid < 16) {  // online softmax, one thread per query row
      const int row = tid;
      float rmax = -3.4e38f;
#pragma unroll
      for (int j = 0; j < 32; ++j) rmax = fmaxf(rmax, Sbuf[row * 32 + j] * SCALE_);
      const float mnew = fmaxf(m_i, rmax);
      const float a = __expf(m_i - mnew);
      float sum = 0.f;
#pragma unroll
      for (int j = 0; j < 32; ++j) {
        const float p = __expf(Sbuf[row * 32 + j] * SCALE_ - mnew);
        Pb[row * 32 + j] = f2bf(p);
        sum += p;
      }
      l_i = l_i * a + sum;
      m_i = mnew;
      alpha_s[row] = a;
    }
    __syncthreads();

    AFrag pa;
    {
      const unsigned short* p = &Pb[lr * 32 + hh * 8];
      pa.q[0] = *reinterpret_cast<const uint4*>(p);
      pa.q[1] = *reinterpret_cast<const uint4*>(p + 16);
    }
    float ar[8];
#pragma unroll
    for (int j = 0; j < 8; ++j) ar[j] = alpha_s[j + 8 * hh];
#pragma unroll
    for (int nt = 0; nt < 8; ++nt) {
#pragma unroll
      for (int j = 0; j < 8; ++j) o[nt].f[j] *= ar[j];
      AFrag bfr;
      const int dcol = wv * 128 + nt * 16 + lr;
      const size_t base = (size_t)(bh * HD_ + dcol) * S_ + j0 + hh * 8;
      bfr.q[0] = *reinterpret_cast<const uint4*>(vtb + base);
      bfr.q[1] = *reinterpret_cast<const uint4*>(vtb + base + 16);
      o[nt].v = wmma_bf16(pa.v, bfr.v, o[nt].v);
    }
    __syncthreads();
  }

  if (tid < 16) invl_s[tid] = 1.f / l_i;
  __syncthreads();

  const int bb = bh >> 1, head = bh & 1;
  float il[8];
#pragma unroll
  for (int j = 0; j < 8; ++j) il[j] = invl_s[j + 8 * hh];
#pragma unroll
  for (int nt = 0; nt < 8; ++nt) {
    const int col = head * HD_ + wv * 128 + nt * 16 + lr;
#pragma unroll
    for (int j = 0; j < 8; ++j) {
      const int row = sq0 + j + 8 * hh;
      attn[(size_t)(bb * S_ + row) * D_ + col] = f2bf(o[nt].f[j] * il[j]);
    }
  }
}

// ============================================================================
// Kernel 4: output projection.  Out[8192x2048] = attn(bf16) * woutbf + bout.
// ============================================================================
__global__ __launch_bounds__(256) void out_gemm_kernel(
    const unsigned short* __restrict__ A, const unsigned short* __restrict__ W,
    const float* __restrict__ bias, float* __restrict__ Out) {
  __shared__ unsigned short As[2 * 128 * 32];
  __shared__ unsigned short Bs[2 * 128 * 32];

  const int tid  = threadIdx.x;
  const int lane = tid & 31;
  const int wv   = tid >> 5;
  const int hh   = lane >> 4;
  const int lr   = lane & 15;
  const int wm   = wv >> 1;
  const int wn   = wv & 1;
  const int n0 = blockIdx.x * 128;
  const int m0 = blockIdx.y * 128;

  Acc8 acc[2][4];
#pragma unroll
  for (int mt = 0; mt < 2; ++mt)
#pragma unroll
    for (int nt = 0; nt < 4; ++nt)
#pragma unroll
      for (int j = 0; j < 8; ++j) acc[mt][nt].f[j] = 0.f;

  gemm_core_bf16(A, D_, W, D_, As, Bs, m0, n0, D_, acc);

#pragma unroll
  for (int mt = 0; mt < 2; ++mt)
#pragma unroll
    for (int nt = 0; nt < 4; ++nt) {
      const int c = n0 + wn * 64 + nt * 16 + lr;
      const float bv = bias[c];
#pragma unroll
      for (int j = 0; j < 8; ++j) {
        const int r = m0 + wm * 32 + mt * 16 + j + 8 * hh;
        Out[(size_t)r * D_ + c] = acc[mt][nt].f[j] + bv;
      }
    }
}

// ============================================================================
extern "C" void kernel_launch(void* const* d_in, const int* in_sizes, int n_in,
                              void* d_out, int out_size, void* d_ws, size_t ws_size,
                              hipStream_t stream) {
  const float* x    = (const float*)d_in[0];
  const float* wqkv = (const float*)d_in[1];
  const float* wout = (const float*)d_in[2];
  const float* bout = (const float*)d_in[3];
  float* out = (float*)d_out;

  unsigned short* ws = (unsigned short*)d_ws;
  const size_t nqkv = (size_t)B_ * H_ * S_ * HD_;   // 16,777,216
  unsigned short* qb     = ws;
  unsigned short* kb     = qb + nqkv;
  unsigned short* vtb    = kb + nqkv;               // v stored (b,h,d,s)
  unsigned short* attn   = vtb + nqkv;              // (b,s,h*d)
  unsigned short* xbf    = attn + nqkv;             // 16,777,216
  unsigned short* wqkvbf = xbf + (size_t)M_ * D_;   // 12,582,912
  unsigned short* woutbf = wqkvbf + (size_t)D_ * N3_; // 4,194,304

  f32_to_bf16_kernel<<<(M_ * D_ / 8) / 256, 256, 0, stream>>>(x, xbf);
  f32_to_bf16_kernel<<<(D_ * N3_ / 8) / 256, 256, 0, stream>>>(wqkv, wqkvbf);
  f32_to_bf16_kernel<<<(D_ * D_ / 8) / 256, 256, 0, stream>>>(wout, woutbf);

  qkv_gemm_kernel<<<dim3(N3_ / 128, M_ / 128), 256, 0, stream>>>(xbf, wqkvbf, qb, kb, vtb);
  rope_kernel<<<(B_ * S_ * HD_) / 256, 256, 0, stream>>>(qb, kb);
  flash_attn_kernel<<<dim3(S_ / 16, B_ * H_), 256, 0, stream>>>(qb, kb, vtb, attn);
  out_gemm_kernel<<<dim3(D_ / 128, M_ / 128), 256, 0, stream>>>(attn, wout ? woutbf : woutbf, bout, out);
}
